// QuantumGateAttention_19516331393142
// MI455X (gfx1250) — compile-verified, matrix operand order checked by reading, not run on hardware
//
#include <hip/hip_runtime.h>
#include <hip/hip_bf16.h>

typedef __attribute__((ext_vector_type(16))) _Float16 v16h;
typedef __attribute__((ext_vector_type(8)))  _Float16 v8h;
typedef __attribute__((ext_vector_type(8)))  float    v8f;

#define D_MODEL 1024
#define N_HEADS 16
#define D_K     64
#define SEQ     2048
#define BATCH   2

// ---------------------------------------------------------------------------
// Optional gfx1250 async global->LDS path (ASYNCcnt-tracked, no VGPR bounce).
// Guarded so compilation cannot regress if the builtins are absent.
// Probe-discovered signature: (v4i addrspace(1)*, v4i addrspace(3)*, Ii, Ii)
// ---------------------------------------------------------------------------
#if __has_builtin(__builtin_amdgcn_global_load_async_to_lds_b128) && \
    __has_builtin(__builtin_amdgcn_s_wait_asynccnt)
#define QGA_ASYNC 1
#else
#define QGA_ASYNC 0
#endif

#if QGA_ASYNC
typedef __attribute__((__vector_size__(4 * sizeof(int)))) int qga_v4i;

__device__ __forceinline__ void async_g2l_b128(const void* g, void* l) {
    __builtin_amdgcn_global_load_async_to_lds_b128(
        (__attribute__((address_space(1))) qga_v4i*)g,
        (__attribute__((address_space(3))) qga_v4i*)l, 0, 0);
}
__device__ __forceinline__ void async_wait0() {
    __builtin_amdgcn_s_wait_asynccnt(0);
}
#else
__device__ __forceinline__ void async_wait0() {}
#endif

// ---------------------------------------------------------------------------
// WMMA helpers
// ---------------------------------------------------------------------------
__device__ __forceinline__ v8f wmma_f16(v16h a, v16h b, v8f c) {
    return __builtin_amdgcn_wmma_f32_16x16x32_f16(false, a, false, b,
                                                  (short)0, c, false, false);
}

__device__ __forceinline__ v8f zero8() {
    v8f z;
#pragma unroll
    for (int i = 0; i < 8; ++i) z[i] = 0.0f;
    return z;
}

// Load a 16x32 f16 A-fragment (or B^T-fragment) from a row-major tile.
// Lane L (h=L>>4, r=L&15) holds row r, K columns {8h..8h+7},{8h+16..8h+23}
// -> two 16B LDS loads per fragment.
__device__ __forceinline__ v16h load_frag(const _Float16* base, int ld) {
    int lane = threadIdx.x & 31;
    int h = lane >> 4;
    int r = lane & 15;
    const _Float16* p = base + r * ld + 8 * h;
    v8h lo = *(const v8h*)(p);
    v8h hi = *(const v8h*)(p + 16);
    return __builtin_shufflevector(lo, hi, 0, 1, 2, 3, 4, 5, 6, 7,
                                   8, 9, 10, 11, 12, 13, 14, 15);
}

// ---------------------------------------------------------------------------
// Kernel 1: scalars  g = sigmoid(mean(gate)); a_h, s_h (with 1/sqrt(dk) fold)
// f[0]=g, f[1..16]=a_h, f[17..32]=s_h
// ---------------------------------------------------------------------------
__global__ void qga_scalars(const float* __restrict__ gate,
                            const float* __restrict__ theta,
                            float* __restrict__ f) {
    __shared__ float red[256];
    int t = threadIdx.x;
    float s = gate[t] + gate[t + 256] + gate[t + 512] + gate[t + 768];
    red[t] = s;
    __syncthreads();
    for (int o = 128; o > 0; o >>= 1) {
        if (t < o) red[t] += red[t + o];
        __syncthreads();
    }
    float g = 1.0f / (1.0f + __expf(-(red[0] * (1.0f / 1024.0f))));
    if (t == 0) f[0] = g;
    if (t < N_HEADS) {
        float th = theta[t];
        f[1 + t]  = ((1.0f - g) + g * cosf(th)) * 0.125f;  // a_h / sqrt(dk)
        f[17 + t] = (g * sinf(th)) * 0.125f;               // s_h / sqrt(dk)
    }
}

// ---------------------------------------------------------------------------
// Kernel 2: weight convert fp32 -> f16.
// mode 0: plain copy. mode 1: fold rotation+gate blend+1/sqrt(dk) into Wq;
// also produce transformed bias (fp32) via c==0 threads.
// ---------------------------------------------------------------------------
__global__ void qga_convert_w(const float* __restrict__ W,
                              const float* __restrict__ b,
                              _Float16* __restrict__ Wout,
                              float* __restrict__ bout,
                              const float* __restrict__ f,
                              int mode) {
    int i = blockIdx.x * 256 + threadIdx.x;     // over 1024*1024
    int r = i >> 10;
    int c = i & 1023;
    float v;
    if (mode == 0) {
        v = W[i];
    } else {
        int idx = r & (D_K - 1);
        int hh  = r >> 6;
        float a = f[1 + hh];
        float s = f[17 + hh];
        if (idx == 0)      v = a * W[i] - s * W[i + D_MODEL];
        else if (idx == 1) v = s * W[i - D_MODEL] + a * W[i];
        else               v = W[i] * 0.125f;
        if (c == 0 && bout != nullptr) {
            float bv;
            if (idx == 0)      bv = a * b[r] - s * b[r + 1];
            else if (idx == 1) bv = s * b[r - 1] + a * b[r];
            else               bv = b[r] * 0.125f;
            bout[r] = bv;
        }
    }
    Wout[i] = (_Float16)v;
}

// ---------------------------------------------------------------------------
// Kernel 3/5: GEMM  Y[4096,1024] = A[4096,1024] @ W^T + bias
// W is f16 row-major [N=1024, K=1024] (== B^T row-major).
// 256 threads (8 waves, 4x2 wave grid), 128x128 tile, K-step 32.
// Each wave: 2 A-frags x 4 B-frags = 8 v_wmma per k-step.
// ---------------------------------------------------------------------------
template <bool AF32, bool OUTF16>
__global__ void __launch_bounds__(256)
qga_gemm128(const void* __restrict__ Ap, const _Float16* __restrict__ W,
            const float* __restrict__ bias, void* __restrict__ Yp) {
    __shared__ _Float16 At[128 * 40];
    __shared__ _Float16 Wt[128 * 40];

    const int t = threadIdx.x;
    const int wv = t >> 5;
    const int lane = t & 31;
    const int h2 = lane >> 4;
    const int r = lane & 15;
    const int mw = wv & 3;         // m sub-tile (32 rows)
    const int nw = wv >> 2;        // n half (64 cols)
    const int m0 = blockIdx.x * 128;
    const int n0 = blockIdx.y * 128;

    v8f acc[2][4];
#pragma unroll
    for (int i = 0; i < 2; ++i)
#pragma unroll
        for (int n = 0; n < 4; ++n) acc[i][n] = zero8();

    for (int k0 = 0; k0 < D_MODEL; k0 += 32) {
        // --- fill A tile: 128 x 32
        if (AF32) {
            const float* A = (const float*)Ap;
#pragma unroll
            for (int p = 0; p < 4; ++p) {
                int lin = p * 1024 + t * 4;
                int row = lin >> 5, col = lin & 31;
                const float4 v =
                    *(const float4*)(A + (size_t)(m0 + row) * D_MODEL + k0 + col);
                At[row * 40 + col + 0] = (_Float16)v.x;
                At[row * 40 + col + 1] = (_Float16)v.y;
                At[row * 40 + col + 2] = (_Float16)v.z;
                At[row * 40 + col + 3] = (_Float16)v.w;
            }
            if (k0 + 32 < D_MODEL)
                __builtin_prefetch(A + (size_t)(m0 + (t >> 1)) * D_MODEL + k0 + 32,
                                   0, 1);
        } else {
            const _Float16* A = (const _Float16*)Ap;
#pragma unroll
            for (int p = 0; p < 2; ++p) {
                int lin = p * 2048 + t * 8;
                int row = lin >> 5, col = lin & 31;
#if QGA_ASYNC
                async_g2l_b128(A + (size_t)(m0 + row) * D_MODEL + k0 + col,
                               &At[row * 40 + col]);
#else
                *(v8h*)(&At[row * 40 + col]) =
                    *(const v8h*)(A + (size_t)(m0 + row) * D_MODEL + k0 + col);
#endif
            }
        }
        // --- fill W tile: 128 x 32
#pragma unroll
        for (int p = 0; p < 2; ++p) {
            int lin = p * 2048 + t * 8;
            int row = lin >> 5, col = lin & 31;
#if QGA_ASYNC
            async_g2l_b128(W + (size_t)(n0 + row) * D_MODEL + k0 + col,
                           &Wt[row * 40 + col]);
#else
            *(v8h*)(&Wt[row * 40 + col]) =
                *(const v8h*)(W + (size_t)(n0 + row) * D_MODEL + k0 + col);
#endif
        }
        async_wait0();
        __syncthreads();

        v16h af[2];
#pragma unroll
        for (int i = 0; i < 2; ++i)
            af[i] = load_frag(&At[(32 * mw + 16 * i) * 40], 40);
#pragma unroll
        for (int n = 0; n < 4; ++n) {
            v16h bf = load_frag(&Wt[(64 * nw + 16 * n) * 40], 40);
#pragma unroll
            for (int i = 0; i < 2; ++i)
                acc[i][n] = wmma_f16(af[i], bf, acc[i][n]);
        }
        __syncthreads();
    }

    // epilogue: C/D layout row = v+8h, col = r
#pragma unroll
    for (int i = 0; i < 2; ++i) {
#pragma unroll
        for (int n = 0; n < 4; ++n) {
            int col = n0 + 64 * nw + 16 * n + r;
            float bv = bias[col];
#pragma unroll
            for (int v = 0; v < 8; ++v) {
                int gm = m0 + 32 * mw + 16 * i + v + 8 * h2;
                float y = acc[i][n][v] + bv;
                if (OUTF16)
                    ((_Float16*)Yp)[(size_t)gm * D_MODEL + col] = (_Float16)y;
                else
                    ((float*)Yp)[(size_t)gm * D_MODEL + col] = y;
            }
        }
    }
}

// ---------------------------------------------------------------------------
// Kernel 4: flash attention. 256 threads (8 waves), 128 q-rows per block,
// 64-row K/V tiles, one (b,head) per blockIdx.y.
// Q already holds Q_eff (rotation+gate+1/sqrt(dk) folded into Wq).
// ---------------------------------------------------------------------------
__global__ void __launch_bounds__(256)
qga_flash_attn(const _Float16* __restrict__ Q, const _Float16* __restrict__ K,
               const _Float16* __restrict__ V, _Float16* __restrict__ O) {
    __shared__ _Float16 Qt[128 * 72];
    __shared__ _Float16 Kt[64 * 72];
    __shared__ _Float16 Vt[64 * 72];   // transposed: [d][kseq]
    __shared__ _Float16 Pt[128 * 72];

    const int t = threadIdx.x;
    const int w = t >> 5;              // wave 0..7 -> q rows 16w..16w+15
    const int lane = t & 31;
    const int h2 = lane >> 4;
    const int r = lane & 15;
    const int q0 = blockIdx.x * 128;
    const int bh = blockIdx.y;
    const int b = bh >> 4;
    const int hd = bh & 15;
    const size_t baseRow = (size_t)b * SEQ;

    // load Q tile once: 128 x 64
#pragma unroll
    for (int p = 0; p < 4; ++p) {
        int lin = p * 2048 + t * 8;
        int row = lin >> 6, col = lin & 63;
#if QGA_ASYNC
        async_g2l_b128(Q + (baseRow + q0 + row) * D_MODEL + hd * D_K + col,
                       &Qt[row * 72 + col]);
#else
        *(v8h*)(&Qt[row * 72 + col]) =
            *(const v8h*)(Q + (baseRow + q0 + row) * D_MODEL + hd * D_K + col);
#endif
    }

    float m_run[8], l_run[8];
    v8f acc[4];
#pragma unroll
    for (int v = 0; v < 8; ++v) { m_run[v] = -3.0e38f; l_run[v] = 0.0f; }
#pragma unroll
    for (int n = 0; n < 4; ++n) acc[n] = zero8();

    for (int k0 = 0; k0 < SEQ; k0 += 64) {
        __syncthreads();  // previous iter's reads done before refill
#pragma unroll
        for (int p = 0; p < 2; ++p) {
            int lin = p * 2048 + t * 8;
            int row = lin >> 6, col = lin & 63;
#if QGA_ASYNC
            async_g2l_b128(K + (baseRow + k0 + row) * D_MODEL + hd * D_K + col,
                           &Kt[row * 72 + col]);
#else
            *(v8h*)(&Kt[row * 72 + col]) =
                *(const v8h*)(K + (baseRow + k0 + row) * D_MODEL + hd * D_K + col);
#endif
            v8h vv =
                *(const v8h*)(V + (baseRow + k0 + row) * D_MODEL + hd * D_K + col);
#pragma unroll
            for (int j = 0; j < 8; ++j) Vt[(col + j) * 72 + row] = vv[j];
        }
        async_wait0();
        __syncthreads();

        // scores: S = Qtile @ Ktile^T  (K tile row-major == B^T row-major)
        v16h qa0 = load_frag(&Qt[16 * w * 72 + 0], 72);
        v16h qa1 = load_frag(&Qt[16 * w * 72 + 32], 72);
        v8f sf[4];
#pragma unroll
        for (int n = 0; n < 4; ++n) {
            v8f c = zero8();
            c = wmma_f16(qa0, load_frag(&Kt[16 * n * 72 + 0], 72), c);
            c = wmma_f16(qa1, load_frag(&Kt[16 * n * 72 + 32], 72), c);
            sf[n] = c;
        }

        // online softmax (rows v+8h of this wave's 16-row slice)
#pragma unroll
        for (int v = 0; v < 8; ++v) {
            float mx = sf[0][v];
#pragma unroll
            for (int n = 1; n < 4; ++n) mx = fmaxf(mx, sf[n][v]);
#pragma unroll
            for (int off = 1; off < 16; off <<= 1)
                mx = fmaxf(mx, __shfl_xor(mx, off, 32));
            float mnew = fmaxf(m_run[v], mx);
            float alpha = __expf(m_run[v] - mnew);
            m_run[v] = mnew;
            float ps = 0.0f;
#pragma unroll
            for (int n = 0; n < 4; ++n) {
                float p = __expf(sf[n][v] - mnew);
                Pt[(16 * w + v + 8 * h2) * 72 + 16 * n + r] = (_Float16)p;
                ps += p;
            }
#pragma unroll
            for (int off = 1; off < 16; off <<= 1)
                ps += __shfl_xor(ps, off, 32);
            l_run[v] = l_run[v] * alpha + ps;
#pragma unroll
            for (int n = 0; n < 4; ++n) acc[n][v] *= alpha;
        }
        __syncthreads();

        // O += P @ Vtile  (Vt transposed -> same row-major fragment gather)
        v16h pa0 = load_frag(&Pt[16 * w * 72 + 0], 72);
        v16h pa1 = load_frag(&Pt[16 * w * 72 + 32], 72);
#pragma unroll
        for (int n = 0; n < 4; ++n) {
            acc[n] = wmma_f16(pa0, load_frag(&Vt[16 * n * 72 + 0], 72), acc[n]);
            acc[n] = wmma_f16(pa1, load_frag(&Vt[16 * n * 72 + 32], 72), acc[n]);
        }
    }

    // write normalized output (f16) back to [B,S,H*dk] layout
#pragma unroll
    for (int n = 0; n < 4; ++n) {
#pragma unroll
        for (int v = 0; v < 8; ++v) {
            size_t row = baseRow + q0 + 16 * w + v + 8 * h2;
            O[row * D_MODEL + hd * D_K + 16 * n + r] =
                (_Float16)(acc[n][v] / l_run[v]);
        }
    }
}

// ---------------------------------------------------------------------------
// Host launcher
// ---------------------------------------------------------------------------
extern "C" void kernel_launch(void* const* d_in, const int* in_sizes, int n_in,
                              void* d_out, int out_size, void* d_ws, size_t ws_size,
                              hipStream_t stream) {
    (void)in_sizes; (void)n_in; (void)out_size; (void)ws_size;

    const float* query = (const float*)d_in[0];
    const float* key_  = (const float*)d_in[1];
    const float* value = (const float*)d_in[2];
    const float* Wq    = (const float*)d_in[3];
    const float* bq    = (const float*)d_in[4];
    const float* Wk    = (const float*)d_in[5];
    const float* bk    = (const float*)d_in[6];
    const float* Wv    = (const float*)d_in[7];
    const float* bv    = (const float*)d_in[8];
    const float* Wo    = (const float*)d_in[9];
    const float* bo    = (const float*)d_in[10];
    const float* theta = (const float*)d_in[11];
    const float* gate  = (const float*)d_in[12];

    // workspace layout: [0,16KB) fp32 scalars+bias, then f16 region (~40MB)
    float* f = (float*)d_ws;                 // f[0]=g, f[1..16]=a, f[17..32]=s
    float* bq_eff = f + 256;                 // 1024 fp32
    _Float16* hb = (_Float16*)((char*)d_ws + 16384);
    const size_t WSZ = (size_t)D_MODEL * D_MODEL;   // 1M elements
    _Float16* wq_e = hb;                 // Wq folded (f16)
    _Float16* wk_h = hb + 1 * WSZ;
    _Float16* wv_h = hb + 2 * WSZ;
    _Float16* wo_h = hb + 3 * WSZ;
    _Float16* qe   = hb + 4 * WSZ;       // Q_eff   [4096,1024]
    _Float16* kh   = hb + 8 * WSZ;       // K proj
    _Float16* vh   = hb + 12 * WSZ;      // V proj
    _Float16* ao   = hb + 16 * WSZ;      // attention output

    qga_scalars<<<1, 256, 0, stream>>>(gate, theta, f);

    qga_convert_w<<<4096, 256, 0, stream>>>(Wq, bq, wq_e, bq_eff, f, 1);
    qga_convert_w<<<4096, 256, 0, stream>>>(Wk, nullptr, wk_h, nullptr, f, 0);
    qga_convert_w<<<4096, 256, 0, stream>>>(Wv, nullptr, wv_h, nullptr, f, 0);
    qga_convert_w<<<4096, 256, 0, stream>>>(Wo, nullptr, wo_h, nullptr, f, 0);

    dim3 ggrid(BATCH * SEQ / 128, D_MODEL / 128);  // (32, 8)
    qga_gemm128<true, true><<<ggrid, 256, 0, stream>>>(query, wq_e, bq_eff, qe);
    qga_gemm128<true, true><<<ggrid, 256, 0, stream>>>(key_, wk_h, bk, kh);
    qga_gemm128<true, true><<<ggrid, 256, 0, stream>>>(value, wv_h, bv, vh);

    dim3 agrid(SEQ / 128, BATCH * N_HEADS);        // (16, 32)
    qga_flash_attn<<<agrid, 256, 0, stream>>>(qe, kh, vh, ao);

    qga_gemm128<false, false><<<ggrid, 256, 0, stream>>>(ao, wo_h, bo,
                                                         (float*)d_out);
}
